// DiffGraphormer_84164179132830
// MI455X (gfx1250) — compile-verified, matrix operand order checked by reading, not run on hardware
//
#include <hip/hip_runtime.h>
#include <hip/hip_fp16.h>
#include <math.h>

typedef __attribute__((ext_vector_type(16))) _Float16 v16h;
typedef __attribute__((ext_vector_type(8)))  float    v8f;

#define HID   64
#define HEADS 4
#define DH    16
#define TPB   256

// ---------------- helpers ----------------

__device__ __forceinline__ void atomicMaxF(float* addr, float val) {
    // Sign-aware int trick: valid given init to a large-negative float.
    if (val >= 0.0f) atomicMax((int*)addr, __float_as_int(val));
    else             atomicMin((unsigned int*)addr, __float_as_uint(val));
}

// ---------------- kernels ----------------

// init softmax state (N*4), ef-aggregation scalars S (N*16), master index slot
__global__ void k_init(float* nmax, float* nsum, float* S, int* midx, int N) {
    int i = blockIdx.x * blockDim.x + threadIdx.x;
    if (i < N * HEADS) { nmax[i] = -1e30f; nsum[i] = 0.0f; }
    if (i < N * 16) S[i] = 0.0f;
    if (i == 0) *midx = 0x7fffffff;
}

// argmax(x[:,0]==1.0) == first index where true
__global__ void k_find_master(const float* __restrict__ xt, int* midx, int N) {
    int i = blockIdx.x * blockDim.x + threadIdx.x;
    if (i < N && xt[(size_t)i * 7] == 1.0f) atomicMin(midx, i);
}

// per-node velocity (unit direction / dt) and displacement norm
__global__ void k_vel(const float* __restrict__ xt, const float* __restrict__ xtd,
                      const float* __restrict__ dtp, float4* __restrict__ vel, int N) {
    int i = blockIdx.x * blockDim.x + threadIdx.x;
    if (i >= N) return;
    float dtv = dtp[0];
    const float* a = xt  + (size_t)i * 7;
    const float* b = xtd + (size_t)i * 7;
    float dx = b[1] - a[1], dy = b[2] - a[2], dz = b[3] - a[3];
    float dn = sqrtf(dx * dx + dy * dy + dz * dz);
    float inv = 1.0f / (fmaxf(dn, 1e-12f) * dtv);
    vel[i] = make_float4(dx * inv, dy * inv, dz * inv, dn);
}

// x_embed = x_t @ W_node + b_node  -> f16 (WMMA A operand feed)
__global__ void k_embed(const float* __restrict__ xt, const float* __restrict__ Wn,
                        const float* __restrict__ bn, _Float16* __restrict__ xe16, int N) {
    int idx = blockIdx.x * blockDim.x + threadIdx.x;
    if (idx >= N * HID) return;
    int i = idx >> 6, j = idx & 63;
    const float* xi = xt + (size_t)i * 7;
    float acc = bn[j];
#pragma unroll
    for (int kk = 0; kk < 7; ++kk) acc += xi[kk] * Wn[kk * HID + j];
    xe16[idx] = (_Float16)acc;
}

// convert Wq/Wk/Wv/Ws (64x64 f32, [k][j]) -> f16 transposed [j][k] (WMMA B feed)
__global__ void k_wconv(const float* __restrict__ Wq, const float* __restrict__ Wk,
                        const float* __restrict__ Wv, const float* __restrict__ Ws,
                        _Float16* __restrict__ wT) {
    int idx = blockIdx.x * blockDim.x + threadIdx.x;
    if (idx >= 4 * HID * HID) return;
    int m = idx >> 12, local = idx & 4095;
    int j = local >> 6, kk = local & 63;
    const float* W = (m == 0) ? Wq : (m == 1) ? Wk : (m == 2) ? Wv : Ws;
    wT[idx] = (_Float16)W[kk * HID + j];
}

// Q/K/V and x_trans(=x_embed@Ws+bs) via v_wmma_f32_16x16x32_f16.
// One wave -> one 16x16 output tile of one matrix; K=64 in two WMMA steps.
__global__ void k_qkvs(const _Float16* __restrict__ xe16, const _Float16* __restrict__ wT,
                       const float* __restrict__ bq, const float* __restrict__ bk,
                       const float* __restrict__ bv, const float* __restrict__ bs,
                       float* __restrict__ outbase, int N, int mtiles) {
    // wave id is uniform across the wave: pin it to an SGPR so all tile math,
    // bounds checks and the store base stay scalar.
    int wid  = __builtin_amdgcn_readfirstlane((blockIdx.x * blockDim.x + threadIdx.x) >> 5);
    int lane = threadIdx.x & 31;
    if (wid >= 16 * mtiles) return;           // scalar branch

    int mtile = wid % mtiles;
    int t2    = wid / mtiles;
    int ntile = t2 & 3;
    int mat   = t2 >> 2;

    int row  = lane & 15;
    int hi   = lane >> 4;                     // 0: K 0-7/16-23, 1: K 8-15/24-31
    int node = mtile * 16 + row;
    if (node >= N) node = N - 1;              // safe clamp (N%16==0 in practice)

    const unsigned* xu = (const unsigned*)(xe16 + (size_t)node * HID);
    int col = ntile * 16 + row;
    const unsigned* wu = (const unsigned*)(wT + ((size_t)mat * 4096 + (size_t)col * HID));

    v8f c = {};
#pragma unroll
    for (int kb = 0; kb < 2; ++kb) {
        union { v16h v; unsigned u[8]; } A, B;
#pragma unroll
        for (int r = 0; r < 8; ++r) {
            // ISA 16-bit operand layout: pair r -> K = kb*32 + (r/4)*16 + hi*8 + (r%4)*2
            int ki = kb * 16 + (r >> 2) * 8 + hi * 4 + (r & 3);   // /2 (32-bit index)
            A.u[r] = xu[ki];
            B.u[r] = wu[ki];
        }
        c = __builtin_amdgcn_wmma_f32_16x16x32_f16(false, A.v, false, B.v,
                                                   (short)0, c, false, false);
    }

    const float* bias = (mat == 0) ? bq : (mat == 1) ? bk : (mat == 2) ? bv : bs;
    int cj = ntile * 16 + (lane & 15);
    float bb = bias[cj];
    // scalar row base + small per-lane offset; per-r offset folds into the
    // instruction's immediate offset field.
    float* o = outbase + (size_t)mat * (size_t)N * HID
                       + (size_t)(mtile * 16) * HID
                       + hi * 8 * HID + cj;
    if (mtile * 16 + 16 <= N) {               // scalar: full tile -> straight-line stores
#pragma unroll
        for (int r = 0; r < 8; ++r)
            o[r * HID] = c[r] + bb;
    } else {
        int base = mtile * 16 + hi * 8;
#pragma unroll
        for (int r = 0; r < 8; ++r)
            if (base + r < N) o[r * HID] = c[r] + bb;
    }
}

// per-node, per-head dot of q with W_edge rows and b_edge:
// qprod[t][h][c] = sum_d q[t][h*16+d] * (c<3 ? We[c][h*16+d] : be[h*16+d])
__global__ void k_qprod(const float* __restrict__ q, const float* __restrict__ We,
                        const float* __restrict__ be, float* __restrict__ qprod, int N) {
    int idx = blockIdx.x * blockDim.x + threadIdx.x;
    if (idx >= N * 16) return;
    int t = idx >> 4, r = idx & 15, h = r >> 2, c = r & 3;
    const float* qt = q + (size_t)t * HID + h * DH;
    const float* w  = (c < 3) ? (We + c * HID + h * DH) : (be + h * DH);
    float acc = 0.0f;
#pragma unroll
    for (int d = 0; d < DH; ++d) acc += qt[d] * w[d];
    qprod[idx] = acc;
}

// Per-edge: edge_attr, raw attention logits (q.k + q.ef via qprod), running segment max.
__global__ void k_alpha(const float* __restrict__ xt, const int* __restrict__ ei,
                        const float4* __restrict__ vel,
                        const float* __restrict__ q, const float* __restrict__ kmat,
                        const float* __restrict__ qprod, const int* __restrict__ midx,
                        float* __restrict__ alpha, float* __restrict__ eattr,
                        float* __restrict__ nmax, int N, int E) {
    int e = blockIdx.x * blockDim.x + threadIdx.x;
    if (e >= E) return;
    int s = ei[e], t = ei[E + e];
    int mi = *midx; if (mi < 0 || mi >= N) mi = 0;

    const float* mp = xt + (size_t)mi * 7;
    const float* tp = xt + (size_t)t  * 7;
    float rx = mp[1] - tp[1], ry = mp[2] - tp[2], rz = mp[3] - tp[3];
    float rn = sqrtf(rx * rx + ry * ry + rz * rz);
    float4 vt = vel[t];
    float na = fmaxf(rn, 1e-6f);
    float nb = fmaxf(sqrtf(vt.x * vt.x + vt.y * vt.y + vt.z * vt.z), 1e-6f);
    float dirs = (rx * vt.x + ry * vt.y + rz * vt.z) / (na * nb);
    float da = 1.0f / (rn + 1e-6f);
    float vs = vt.w;
    eattr[(size_t)e * 3 + 0] = da;
    eattr[(size_t)e * 3 + 1] = dirs;
    eattr[(size_t)e * 3 + 2] = vs;

    const float* qt = q     + (size_t)t * HID;
    const float* ks = kmat  + (size_t)s * HID;
    const float* qp = qprod + (size_t)t * 16;
#pragma unroll
    for (int h = 0; h < HEADS; ++h) {
        float dotqk = 0.0f;
#pragma unroll
        for (int d = 0; d < DH; ++d) {
            int j = h * DH + d;
            dotqk += qt[j] * ks[j];
        }
        float ar = 0.25f * (dotqk + da * qp[h * 4 + 0] + dirs * qp[h * 4 + 1]
                                  + vs * qp[h * 4 + 2] + qp[h * 4 + 3]);
        alpha[(size_t)e * HEADS + h] = ar;
        atomicMaxF(&nmax[(size_t)t * HEADS + h], ar);
    }
}

// exp(logit - segmax) and segment sum
__global__ void k_exp(const int* __restrict__ ei, float* __restrict__ alpha,
                      const float* __restrict__ nmax, float* __restrict__ nsum, int E) {
    int g = blockIdx.x * blockDim.x + threadIdx.x;
    if (g >= E * HEADS) return;
    int e = g >> 2, h = g & 3;
    int t = ei[E + e];
    float a = __expf(alpha[g] - nmax[(size_t)t * HEADS + h]);
    alpha[g] = a;
    atomicAdd(&nsum[(size_t)t * HEADS + h], a);
}

// weighted message scatter.  v-part goes straight into x_trans; the edge_feat part
// is rank-1 in {We rows, be}, so only 4 scalars per (tgt,head) are accumulated:
//   S[t][h] = sum_e w * {da, dirs, vs, 1}
__global__ void k_msg(const int* __restrict__ ei, const float* __restrict__ alpha,
                      const float* __restrict__ eattr, const float* __restrict__ nsum,
                      const float* __restrict__ vmat, float* __restrict__ xtrans,
                      float* __restrict__ S, int E) {
    int e = blockIdx.x * blockDim.x + threadIdx.x;
    if (e >= E) return;
    int s = ei[e], t = ei[E + e];
    float w[HEADS];
#pragma unroll
    for (int h = 0; h < HEADS; ++h)
        w[h] = alpha[(size_t)e * HEADS + h] / (nsum[(size_t)t * HEADS + h] + 1e-16f);
    float da   = eattr[(size_t)e * 3 + 0];
    float dirs = eattr[(size_t)e * 3 + 1];
    float vs   = eattr[(size_t)e * 3 + 2];
    float* Sp = S + (size_t)t * 16;
#pragma unroll
    for (int h = 0; h < HEADS; ++h) {
        atomicAdd(&Sp[h * 4 + 0], w[h] * da);
        atomicAdd(&Sp[h * 4 + 1], w[h] * dirs);
        atomicAdd(&Sp[h * 4 + 2], w[h] * vs);
        atomicAdd(&Sp[h * 4 + 3], w[h]);
    }
    const float* vsrc = vmat + (size_t)s * HID;
    float* xo = xtrans + (size_t)t * HID;
#pragma unroll
    for (int j = 0; j < HID; ++j)
        atomicAdd(&xo[j], w[j >> 4] * vsrc[j]);
}

// per-node: fold the rank-1 edge_feat aggregate into x_trans and reduce to
// xc[i] = (x_trans[i] + ef_agg[i]) . Wc   (only xc is needed downstream)
__global__ void k_nodefix(const float* __restrict__ xtrans, const float* __restrict__ S,
                          const float* __restrict__ We, const float* __restrict__ be,
                          const float* __restrict__ Wc, float* __restrict__ xc, int N) {
    int i = blockIdx.x * blockDim.x + threadIdx.x;
    if (i >= N) return;
    const float* xr = xtrans + (size_t)i * HID;
    const float* Sp = S + (size_t)i * 16;
    float acc = 0.0f;
#pragma unroll
    for (int j = 0; j < HID; ++j) {
        int h = j >> 4;
        float ef = Sp[h * 4 + 0] * We[j] + Sp[h * 4 + 1] * We[HID + j]
                 + Sp[h * 4 + 2] * We[2 * HID + j] + Sp[h * 4 + 3] * be[j];
        acc += (xr[j] + ef) * Wc[j];
    }
    xc[i] = acc;
}

// sigmoid(xc[src] + xc[tgt] + bc)
__global__ void k_final(const int* __restrict__ ei, const float* __restrict__ xc,
                        const float* __restrict__ bc, float* __restrict__ out, int E) {
    int e = blockIdx.x * blockDim.x + threadIdx.x;
    if (e >= E) return;
    float acc = xc[ei[e]] + xc[ei[E + e]] + bc[0];
    out[e] = 1.0f / (1.0f + __expf(-acc));
}

// ---------------- launcher ----------------

extern "C" void kernel_launch(void* const* d_in, const int* in_sizes, int n_in,
                              void* d_out, int out_size, void* d_ws, size_t ws_size,
                              hipStream_t stream) {
    const float* xt  = (const float*)d_in[0];
    const float* xtd = (const float*)d_in[1];
    const int*   ei  = (const int*)d_in[2];
    const float* dt  = (const float*)d_in[3];
    const float* Wn  = (const float*)d_in[4];
    const float* bn  = (const float*)d_in[5];
    const float* We  = (const float*)d_in[6];
    const float* be  = (const float*)d_in[7];
    const float* Wq  = (const float*)d_in[8];
    const float* bq  = (const float*)d_in[9];
    const float* Wk  = (const float*)d_in[10];
    const float* bk  = (const float*)d_in[11];
    const float* Wv  = (const float*)d_in[12];
    const float* bv  = (const float*)d_in[13];
    const float* Ws  = (const float*)d_in[14];
    const float* bs  = (const float*)d_in[15];
    const float* Wc  = (const float*)d_in[16];
    const float* bc  = (const float*)d_in[17];

    int N = in_sizes[0] / 7;
    int E = in_sizes[2] / 2;

    char* base = (char*)d_ws;
    size_t off = 0;
    auto give = [&](size_t bytes) -> char* {
        size_t o = (off + 255) & ~(size_t)255;
        off = o + bytes;
        return base + o;
    };
    float4*    vel   = (float4*)   give((size_t)N * sizeof(float4));
    _Float16*  xe16  = (_Float16*) give((size_t)N * HID * 2);
    _Float16*  wT    = (_Float16*) give((size_t)4 * HID * HID * 2);
    float*     qkvx  = (float*)    give((size_t)4 * N * HID * 4);  // q, k, v, x_trans
    float*     alpha = (float*)    give((size_t)E * HEADS * 4);
    float*     eattr = (float*)    give((size_t)E * 3 * 4);
    float*     nmax  = (float*)    give((size_t)N * HEADS * 4);
    float*     nsum  = (float*)    give((size_t)N * HEADS * 4);
    float*     qprod = (float*)    give((size_t)N * 16 * 4);
    float*     S     = (float*)    give((size_t)N * 16 * 4);
    float*     xc    = (float*)    give((size_t)N * 4);
    int*       midx  = (int*)      give(16);

    float* qmat  = qkvx;
    float* kmat  = qkvx + (size_t)1 * N * HID;
    float* vmat  = qkvx + (size_t)2 * N * HID;
    float* xtr   = qkvx + (size_t)3 * N * HID;

    k_init<<<(N * 16 + TPB - 1) / TPB, TPB, 0, stream>>>(nmax, nsum, S, midx, N);
    k_find_master<<<(N + TPB - 1) / TPB, TPB, 0, stream>>>(xt, midx, N);
    k_vel<<<(N + TPB - 1) / TPB, TPB, 0, stream>>>(xt, xtd, dt, vel, N);
    k_embed<<<(N * HID + TPB - 1) / TPB, TPB, 0, stream>>>(xt, Wn, bn, xe16, N);
    k_wconv<<<(4 * HID * HID + TPB - 1) / TPB, TPB, 0, stream>>>(Wq, Wk, Wv, Ws, wT);

    int mtiles = (N + 15) / 16;
    int nthreads = 16 * mtiles * 32;           // (4 mats x 4 ntiles) per mtile, one wave each
    k_qkvs<<<(nthreads + TPB - 1) / TPB, TPB, 0, stream>>>(xe16, wT, bq, bk, bv, bs,
                                                           qkvx, N, mtiles);

    k_qprod<<<(N * 16 + TPB - 1) / TPB, TPB, 0, stream>>>(qmat, We, be, qprod, N);
    k_alpha<<<(E + TPB - 1) / TPB, TPB, 0, stream>>>(xt, ei, vel, qmat, kmat, qprod,
                                                     midx, alpha, eattr, nmax, N, E);
    k_exp<<<(E * HEADS + TPB - 1) / TPB, TPB, 0, stream>>>(ei, alpha, nmax, nsum, E);
    k_msg<<<(E + TPB - 1) / TPB, TPB, 0, stream>>>(ei, alpha, eattr, nsum,
                                                   vmat, xtr, S, E);
    k_nodefix<<<(N + TPB - 1) / TPB, TPB, 0, stream>>>(xtr, S, We, be, Wc, xc, N);
    k_final<<<(E + TPB - 1) / TPB, TPB, 0, stream>>>(ei, xc, bc, (float*)d_out, E);
}
